// Transolver2CrossAttention_ShareKV_56916906607000
// MI455X (gfx1250) — matrix operand order repaired and checked by reading, not compiled, not run on hardware
//
#include <hip/hip_runtime.h>
#include <stdint.h>

// ---------------- problem constants ----------------
#define B_   2
#define N_   32768
#define D_   256
#define M_   512
#define R_   16
#define H_   8
#define DH_  32
#define HD_  256   // H*DH
#define MR_  528   // M + R
#define TN_  64    // rows per tile
#define MT_  64    // logits column chunk

// ---------------- workspace layout (float offsets) ----------------
#define OFF_QACC  ((size_t)0)                 // B*M*D
#define OFF_KVACC ((size_t)262144)            // B*M*D
#define OFF_SUMQ  ((size_t)524288)            // B*M
#define OFF_SUMKV ((size_t)525312)            // B*M
#define OFF_Q     ((size_t)526336)            // B*MR*HD
#define OFF_K     ((size_t)796672)            // B*M*HD
#define OFF_V     ((size_t)1058816)           // B*M*HD
#define OFF_OH    ((size_t)1320960)           // B*MR*HD
#define OFF_END   ((size_t)1591296)           // then B*D*M bf16 (y_proj, transposed)
#define ZERO_CNT  526336

typedef __attribute__((ext_vector_type(16))) __bf16   v16bf;
typedef __attribute__((ext_vector_type(8)))  __bf16   v8bf;
typedef __attribute__((ext_vector_type(8)))  float    v8f;
typedef __attribute__((ext_vector_type(4)))  unsigned v4u;
typedef __attribute__((ext_vector_type(8)))  int      v8i;
typedef __attribute__((ext_vector_type(4)))  int      v4i;

#if defined(__has_builtin)
#if __has_builtin(__builtin_amdgcn_tensor_load_to_lds)
#define HAVE_TDM 1
#endif
#endif

// ---------------- TDM: 1-D f32 tile -> LDS (6-arg clang-23 form) --------------
#ifdef HAVE_TDM
__device__ __forceinline__ void tdm_load_f32(const void* g, unsigned lds_byte_off,
                                             unsigned nelem) {
  uint64_t ga = (uint64_t)(uintptr_t)g;
  v4u g0;
  g0[0] = 1u;                                   // count=1, user descriptor
  g0[1] = lds_byte_off;                         // lds_addr [63:32]
  g0[2] = (unsigned)(ga & 0xFFFFFFFFu);         // global_addr lo
  g0[3] = (unsigned)((ga >> 32) & 0x01FFFFFFu)  // global_addr hi (57-bit)
        | (2u << 30);                           // type=2 ("image")
  v8i g1;
  g1[0] = (int)(2u << 16);                              // wg_mask=0, data_size=4B
  g1[1] = (int)((nelem & 0xFFFFu) << 16);               // tensor_dim0 lo16
  g1[2] = (int)(((nelem >> 16) & 0xFFFFu) | (1u << 16));// tensor_dim0 hi | tensor_dim1=1
  g1[3] = (int)((nelem & 0xFFFFu) << 16);               // tile_dim0 (<=65535)
  g1[4] = 1;                                            // tile_dim1=1, tile_dim2=0
  g1[5] = (int)nelem;                                   // tensor_dim0_stride lo32
  g1[6] = 0;
  g1[7] = 0;
  v4i z4 = {0, 0, 0, 0};                                // D# groups 2/3 unused (<=2-D)
  v8i z8 = {0, 0, 0, 0, 0, 0, 0, 0};
  __builtin_amdgcn_tensor_load_to_lds(g0, g1, z4, z4, z8, 0);
}
#endif

// ---------------- WMMA fragment helpers (all sources fragment-contiguous) -----
__device__ __forceinline__ v8bf ld8(const __bf16* p) { return *(const v8bf*)p; }
__device__ __forceinline__ v16bf mk16(v8bf lo, v8bf hi) {
  return __builtin_shufflevector(lo, hi, 0, 1, 2, 3, 4, 5, 6, 7,
                                         8, 9, 10, 11, 12, 13, 14, 15);
}
// A 16x32, row-major source s[row*ld + k]: halves K{kb..kb+7} and K{kb+16..kb+23}
__device__ __forceinline__ v16bf ldA_rm(const __bf16* s, int ld, int r0, int k0) {
  int lane = threadIdx.x & 31;
  const __bf16* p = s + (r0 + (lane & 15)) * ld + k0 + ((lane >> 4) << 3);
  return mk16(ld8(p), ld8(p + 16));
}
// B 32x16 from N-major storage st[col*ldk + k]: halves K{kb..kb+7} and K{kb+16..kb+23}
__device__ __forceinline__ v16bf ldB_nm(const __bf16* st, int ldk, int k0, int c0) {
  int lane = threadIdx.x & 31;
  const __bf16* p = st + (c0 + (lane & 15)) * ldk + k0 + ((lane >> 4) << 3);
  return mk16(ld8(p), ld8(p + 16));
}
__device__ __forceinline__ v8f wmma_bf16(v16bf a, v16bf b, v8f c) {
  return __builtin_amdgcn_wmma_f32_16x16x32_bf16(false, a, false, b, (short)0, c,
                                                 false, false);
}

// ---------------- shared routine: w tile = softmax(X @ Wsl) -------------------
// Lw : bf16[TN_][M_] row-major logits (overlaps the f32 staging area Xf)
// Xb : bf16[TN_][D_]   (A operand, logits GEMM)
// WbT: bf16[MT_][D_]   (B operand, N-major chunk of W_slice)
// XbT: bf16[D_][TN_]   optional (B operand for w^T X), written during convert
// LwT: bf16[M_][TN_]   optional; if non-null softmax writes w transposed there,
//                      else w is written row-major in place over Lw.
__device__ void compute_w_tile(const float* __restrict__ Xg,
                               const float* __restrict__ Wsl,
                               __bf16* Lw, float* Xf, __bf16* Xb, __bf16* WbT,
                               __bf16* XbT, __bf16* LwT) {
  const int tid  = threadIdx.x;
  const int wave = tid >> 5;
  const int lane = tid & 31;

#ifdef HAVE_TDM
  if (tid < 32) {                                   // wave 0 issues the DMA
    tdm_load_f32(Xg, (unsigned)(uintptr_t)Xf, TN_ * D_);
    __builtin_amdgcn_s_wait_tensorcnt(0);
  }
  __syncthreads();
#else
  for (int i = tid; i < TN_ * D_; i += 256) Xf[i] = Xg[i];
  __syncthreads();
#endif
  for (int i = tid; i < TN_ * D_; i += 256) {
    float v = Xf[i];
    Xb[i] = (__bf16)v;
    if (XbT) XbT[(i & (D_ - 1)) * TN_ + (i >> 8)] = (__bf16)v;
  }
  __syncthreads();

  // logits (TN_ x M_) = X (TN_ x D_) @ Wsl (D_ x M_), MT_-wide column chunks
  for (int mt = 0; mt < M_; mt += MT_) {
    for (int i = tid; i < D_ * MT_; i += 256) {
      int rr = i / MT_;           // d index  (coalesced global read over cc)
      int cc = i % MT_;           // mode within chunk
      WbT[cc * D_ + rr] = (__bf16)Wsl[rr * M_ + mt + cc];
    }
    __syncthreads();
    // (TN_/16) x (MT_/16) = 4 x 4 = 16 tile pairs over 8 waves
    for (int p = wave; p < (TN_ / 16) * (MT_ / 16); p += 8) {
      int rt = p >> 2, ct = p & 3;
      v8f c = {0.f, 0.f, 0.f, 0.f, 0.f, 0.f, 0.f, 0.f};
      for (int k = 0; k < D_; k += 32)
        c = wmma_bf16(ldA_rm(Xb, D_, rt * 16, k), ldB_nm(WbT, D_, k, ct * 16), c);
      int r  = rt * 16 + ((lane >> 4) << 3);
      int cc = mt + ct * 16 + (lane & 15);
      for (int i = 0; i < 8; ++i) Lw[(r + i) * M_ + cc] = (__bf16)c[i];
    }
    __syncthreads();
  }

  // row softmax over M_ = 512 (one wave per row, 16 values per lane)
  for (int r = wave; r < TN_; r += 8) {
    float v[16];
    float mx = -3.0e38f;
    for (int j = 0; j < 16; ++j) {
      v[j] = (float)Lw[r * M_ + lane + 32 * j];
      mx = fmaxf(mx, v[j]);
    }
    for (int o = 16; o >= 1; o >>= 1) mx = fmaxf(mx, __shfl_xor(mx, o, 32));
    float s = 0.f;
    for (int j = 0; j < 16; ++j) { v[j] = __expf(v[j] - mx); s += v[j]; }
    for (int o = 16; o >= 1; o >>= 1) s += __shfl_xor(s, o, 32);
    float inv = 1.f / s;
    if (LwT) {
      for (int j = 0; j < 16; ++j)
        LwT[(lane + 32 * j) * TN_ + r] = (__bf16)(v[j] * inv);
    } else {
      for (int j = 0; j < 16; ++j)
        Lw[r * M_ + lane + 32 * j] = (__bf16)(v[j] * inv);
    }
  }
  __syncthreads();
}

// ---------------- kernel 0: zero accumulators ----------------
__global__ void k_zero(float* __restrict__ ws) {
  size_t i = (size_t)blockIdx.x * 256 + threadIdx.x;
  if (i < ZERO_CNT) ws[i] = 0.f;
}

// ---------------- kernel 1: fused slice-softmax + w^T X accumulation ----------
__global__ __launch_bounds__(256) void k_slice_project(
    const float* __restrict__ qx, const float* __restrict__ kvx,
    const float* __restrict__ Wslq, const float* __restrict__ Wslkv,
    float* __restrict__ ws) {
  __shared__ __align__(16) char smem[229376];      // 224KB (ISA allows 320KB/WG)
  __bf16* Lw  = (__bf16*)smem;                     // [64][512] logits
  float*  Xf  = (float*)smem;                      // f32 staging (overlap, 64KB)
  __bf16* Xb  = (__bf16*)(smem + 65536);           // [64][256]
  __bf16* XbT = (__bf16*)(smem + 98304);           // [256][64]
  __bf16* WbT = (__bf16*)(smem + 131072);          // [64][256] N-major chunk
  __bf16* LwT = (__bf16*)(smem + 163840);          // [512][64] w transposed

  const int side = blockIdx.z;               // 0 = q, 1 = kv
  const int b    = blockIdx.y;
  const int n0   = blockIdx.x * TN_;
  const float* X    = side ? kvx : qx;
  const float* Wsl  = side ? Wslkv : Wslq;
  float* acc  = ws + (side ? OFF_KVACC : OFF_QACC) + (size_t)b * M_ * D_;
  float* sumw = ws + (side ? OFF_SUMKV : OFF_SUMQ) + (size_t)b * M_;

  compute_w_tile(X + ((size_t)b * N_ + n0) * D_, Wsl, Lw, Xf, Xb, WbT, XbT, LwT);

  const int tid = threadIdx.x, wave = tid >> 5, lane = tid & 31;

  // per-mode column sums of w (contiguous rows of LwT)
  for (int m = tid; m < M_; m += 256) {
    float s = 0.f;
    for (int r = 0; r < TN_; ++r) s += (float)LwT[m * TN_ + r];
    unsafeAtomicAdd(&sumw[m], s);
  }

  // acc[m][d] += sum_n w[n][m] * X[n][d]   (K = TN_ = 64, split-K via atomics)
  for (int p = wave; p < (M_ / 16) * (D_ / 16); p += 8) {
    int mtile = p >> 4, dtile = p & 15;
    v8f c = {0.f, 0.f, 0.f, 0.f, 0.f, 0.f, 0.f, 0.f};
    for (int k = 0; k < TN_; k += 32)
      c = wmma_bf16(ldA_rm(LwT, TN_, mtile * 16, k), ldB_nm(XbT, TN_, k, dtile * 16), c);
    int r  = mtile * 16 + ((lane >> 4) << 3);
    int cc = dtile * 16 + (lane & 15);
    for (int i = 0; i < 8; ++i)
      unsafeAtomicAdd(&acc[(size_t)(r + i) * D_ + cc], c[i]);
  }
}

// ---------------- kernel 2a: normalize + LayerNorm + Q/K/V projections --------
__global__ __launch_bounds__(256) void k_ln_qkv(
    float* __restrict__ ws, const float* __restrict__ reg_qx,
    const float* __restrict__ lnqg, const float* __restrict__ lnqb,
    const float* __restrict__ lnkg, const float* __restrict__ lnkb,
    const float* __restrict__ Wq, const float* __restrict__ Wk,
    const float* __restrict__ Wv) {
  __shared__ float lnrow[D_];
  __shared__ float red[8];
  const int bid = blockIdx.x, t = threadIdx.x, wave = t >> 5, lane = t & 31;
  const bool qside = bid < B_ * MR_;
  int b, j;
  if (qside) { b = bid / MR_; j = bid - b * MR_; }
  else       { int r2 = bid - B_ * MR_; b = r2 / M_; j = r2 - b * M_; }

  float val;
  if (qside) {
    if (j < M_)
      val = ws[OFF_QACC + ((size_t)(b * M_ + j)) * D_ + t] /
            ws[OFF_SUMQ + (size_t)b * M_ + j];
    else
      val = reg_qx[((size_t)(b * R_ + (j - M_))) * D_ + t];
  } else {
    val = ws[OFF_KVACC + ((size_t)(b * M_ + j)) * D_ + t] /
          ws[OFF_SUMKV + (size_t)b * M_ + j];
  }

  float v = val;
  for (int o = 16; o >= 1; o >>= 1) v += __shfl_xor(v, o, 32);
  if (lane == 0) red[wave] = v;
  __syncthreads();
  float mu = 0.f;
  for (int i = 0; i < 8; ++i) mu += red[i];
  mu *= (1.f / D_);
  __syncthreads();

  float dv = val - mu;
  v = dv * dv;
  for (int o = 16; o >= 1; o >>= 1) v += __shfl_xor(v, o, 32);
  if (lane == 0) red[wave] = v;
  __syncthreads();
  float var = 0.f;
  for (int i = 0; i < 8; ++i) var += red[i];
  var *= (1.f / D_);

  const float* g  = qside ? lnqg : lnkg;
  const float* be = qside ? lnqb : lnkb;
  lnrow[t] = dv * rsqrtf(var + 1e-5f) * g[t] + be[t];
  __syncthreads();

  if (qside) {
    float a = 0.f;
    for (int k = 0; k < D_; ++k) a += lnrow[k] * Wq[k * HD_ + t];
    ws[OFF_Q + ((size_t)(b * MR_ + j)) * HD_ + t] = a;
  } else {
    float ak = 0.f, av = 0.f;
    for (int k = 0; k < D_; ++k) {
      float l = lnrow[k];
      ak += l * Wk[k * HD_ + t];
      av += l * Wv[k * HD_ + t];
    }
    ws[OFF_K + ((size_t)(b * M_ + j)) * HD_ + t] = ak;
    ws[OFF_V + ((size_t)(b * M_ + j)) * HD_ + t] = av;
  }
}

// ---------------- kernel 2b: attention per (b, h) -----------------------------
__global__ __launch_bounds__(256) void k_attention(float* __restrict__ ws) {
  __shared__ __bf16 Kh[M_ * DH_];   // 32KB
  __shared__ __bf16 Vh[M_ * DH_];   // 32KB
  const int h = blockIdx.x, b = blockIdx.y;
  const int t = threadIdx.x, wave = t >> 5, lane = t & 31;
  const float* kb_ = ws + OFF_K + (size_t)b * M_ * HD_;
  const float* vb_ = ws + OFF_V + (size_t)b * M_ * HD_;
  for (int i = t; i < M_ * DH_; i += 256) {
    int kk = i >> 5, d = i & 31;
    Kh[i] = (__bf16)kb_[kk * HD_ + h * DH_ + d];
    Vh[i] = (__bf16)vb_[kk * HD_ + h * DH_ + d];
  }
  __syncthreads();

  const float scale = 0.17677669529663687f;  // 1/sqrt(DH)
  for (int j = wave; j < MR_; j += 8) {      // waves process rows independently
    float qd = ws[OFF_Q + ((size_t)(b * MR_ + j)) * HD_ + h * DH_ + lane];
    float sc[16];
    for (int i = 0; i < 16; ++i) sc[i] = 0.f;
    for (int d = 0; d < DH_; ++d) {
      float qv = __shfl(qd, d, 32);
      for (int i = 0; i < 16; ++i) sc[i] += qv * (float)Kh[(lane + 32 * i) * DH_ + d];
    }
    float mx = -3.0e38f;
    for (int i = 0; i < 16; ++i) { sc[i] *= scale; mx = fmaxf(mx, sc[i]); }
    for (int o = 16; o >= 1; o >>= 1) mx = fmaxf(mx, __shfl_xor(mx, o, 32));
    float s = 0.f;
    for (int i = 0; i < 16; ++i) { sc[i] = __expf(sc[i] - mx); s += sc[i]; }
    for (int o = 16; o >= 1; o >>= 1) s += __shfl_xor(s, o, 32);
    float inv = 1.f / s;

    float op[DH_];
    for (int d = 0; d < DH_; ++d) op[d] = 0.f;
    for (int i = 0; i < 16; ++i) {
      float aw = sc[i] * inv;
      int key = lane + 32 * i;
      for (int d = 0; d < DH_; ++d) op[d] += aw * (float)Vh[key * DH_ + d];
    }
    float myo = 0.f;
    for (int d = 0; d < DH_; ++d) {
      float vv = op[d];
      for (int o = 16; o >= 1; o >>= 1) vv += __shfl_xor(vv, o, 32);
      if (lane == d) myo = vv;
    }
    ws[OFF_OH + ((size_t)(b * MR_ + j)) * HD_ + h * DH_ + lane] = myo;
  }
}

// ---------------- kernel 2c: output projection (Wo); y_proj stored transposed -
__global__ __launch_bounds__(256) void k_out_proj(float* __restrict__ ws,
                                                  const float* __restrict__ Wo,
                                                  float* __restrict__ out) {
  __shared__ float orow[HD_];
  const int bid = blockIdx.x;
  const int b = bid / MR_, j = bid - b * MR_;
  const int t = threadIdx.x;
  orow[t] = ws[OFF_OH + ((size_t)(b * MR_ + j)) * HD_ + t];
  __syncthreads();
  float a = 0.f;
  for (int k = 0; k < HD_; ++k) a += orow[k] * Wo[k * D_ + t];
  if (j < M_) {
    __bf16* ypt = (__bf16*)(ws + OFF_END);           // [B][D][M] (d-major)
    ypt[((size_t)b * D_ + t) * M_ + j] = (__bf16)a;
  } else {
    out[(size_t)B_ * N_ * D_ + ((size_t)(b * R_ + (j - M_))) * D_ + t] = a;
  }
}

// ---------------- kernel 3: y_recon = softmax(qx@Wsl_q) @ y_proj --------------
__global__ __launch_bounds__(256) void k_recon(const float* __restrict__ qx,
                                               const float* __restrict__ Wslq,
                                               const float* __restrict__ ws,
                                               float* __restrict__ out) {
  __shared__ __align__(16) char smem[131072];      // 128KB
  __bf16* Lw  = (__bf16*)smem;                     // [64][512]
  float*  Xf  = (float*)smem;
  __bf16* Xb  = (__bf16*)(smem + 65536);           // [64][256]
  __bf16* WbT = (__bf16*)(smem + 98304);           // [64][256]
  const int b  = blockIdx.y;
  const int n0 = blockIdx.x * TN_;

  // w written row-major in place (LwT == nullptr); XbT not needed here
  compute_w_tile(qx + ((size_t)b * N_ + n0) * D_, Wslq, Lw, Xf, Xb, WbT,
                 nullptr, nullptr);

  const int wave = threadIdx.x >> 5, lane = threadIdx.x & 31;
  const __bf16* ypt = (const __bf16*)(ws + OFF_END) + (size_t)b * D_ * M_;

  // out tile (TN_ x D_) = w (TN_ x M_) @ y_proj (M_ x D_); 4 x 16 tiles
  for (int p = wave; p < (TN_ / 16) * 16; p += 8) {
    int ntile = p >> 4, dtile = p & 15;
    v8f c = {0.f, 0.f, 0.f, 0.f, 0.f, 0.f, 0.f, 0.f};
    for (int kt = 0; kt < M_; kt += 32)
      c = wmma_bf16(ldA_rm(Lw, M_, ntile * 16, kt), ldB_nm(ypt, M_, kt, dtile * 16), c);
    int r  = n0 + ntile * 16 + ((lane >> 4) << 3);
    int cc = dtile * 16 + (lane & 15);
    for (int i = 0; i < 8; ++i)
      out[((size_t)b * N_ + r + i) * D_ + cc] = c[i];
  }
}

// ---------------- host launcher ----------------
extern "C" void kernel_launch(void* const* d_in, const int* in_sizes, int n_in,
                              void* d_out, int out_size, void* d_ws, size_t ws_size,
                              hipStream_t stream) {
  const float* qx     = (const float*)d_in[0];
  const float* kvx    = (const float*)d_in[1];
  // d_in[2] (x) and d_in[4] (reg_tok_x) are dead in the reference
  const float* reg_qx = (const float*)d_in[3];
  const float* Wslq   = (const float*)d_in[5];
  const float* Wslkv  = (const float*)d_in[6];
  const float* lnqg   = (const float*)d_in[7];
  const float* lnqb   = (const float*)d_in[8];
  const float* lnkg   = (const float*)d_in[9];
  const float* lnkb   = (const float*)d_in[10];
  const float* Wq     = (const float*)d_in[11];
  const float* Wk     = (const float*)d_in[12];
  const float* Wv     = (const float*)d_in[13];
  const float* Wo     = (const float*)d_in[14];
  float* ws  = (float*)d_ws;
  float* out = (float*)d_out;

  k_zero<<<(ZERO_CNT + 255) / 256, 256, 0, stream>>>(ws);
  k_slice_project<<<dim3(N_ / TN_, B_, 2), 256, 0, stream>>>(qx, kvx, Wslq, Wslkv, ws);
  k_ln_qkv<<<B_ * MR_ + B_ * M_, 256, 0, stream>>>(ws, reg_qx, lnqg, lnqb, lnkg, lnkb,
                                                   Wq, Wk, Wv);
  k_attention<<<dim3(H_, B_), 256, 0, stream>>>(ws);
  k_out_proj<<<B_ * MR_, 256, 0, stream>>>(ws, Wo, out);
  k_recon<<<dim3(N_ / TN_, B_), 256, 0, stream>>>(qx, Wslq, ws, out);
}
